// MultiPathwayBilateralSNN_68186900791558
// MI455X (gfx1250) — matrix operand order repaired and checked by reading, not compiled, not run on hardware
//
#include <hip/hip_runtime.h>

typedef __attribute__((ext_vector_type(16))) _Float16 v16h;
typedef __attribute__((ext_vector_type(8)))  _Float16 h8;
typedef __attribute__((ext_vector_type(8)))  float    v8f;

#define TSTEPS 1000
#define BATCH  4096

__device__ __forceinline__ int laneid() { return (int)(threadIdx.x & 31u); }

__device__ __forceinline__ v8f zero8f() {
  v8f z;
#pragma unroll
  for (int i = 0; i < 8; ++i) z[i] = 0.0f;
  return z;
}

__device__ __forceinline__ v16h zero16h() {
  v16h z;
#pragma unroll
  for (int i = 0; i < 16; ++i) z[i] = (_Float16)0.0f;
  return z;
}

__device__ __forceinline__ h8 zero8h() {
  h8 z;
#pragma unroll
  for (int i = 0; i < 8; ++i) z[i] = (_Float16)0.0f;
  return z;
}

__device__ __forceinline__ v8f wmma16(v16h a, v16h b, v8f c) {
  // D = A(16x32) x B(32x16) + C, f16 in / f32 out
  return __builtin_amdgcn_wmma_f32_16x16x32_f16(false, a, false, b, (short)0, c,
                                                false, false);
}

// Build B-matrix (32x16 = K x N) register image from row-major W[fan_out][fan_in].
// B[k][n] = W[nbase+n][kbase+k].  ISA layout: lane half h, element e holds
// K = 8h+e for e<8 and K = 16+8h+(e-8) for e>=8.
__device__ v16h makeB(const float* W, int fan_out, int fan_in, int nbase, int kbase) {
  const int l = laneid();
  const int n = (l & 15) + nbase;
  const int kh = (l >> 4) * 8;
  v16h b = zero16h();
  if (n < fan_out) {
#pragma unroll
    for (int e = 0; e < 16; ++e) {
      int k = kbase + kh + e + ((e < 8) ? 0 : 8);
      if (k < fan_in) b[e] = (_Float16)W[n * fan_in + k];
    }
  }
  return b;
}

// LIF (snntorch Leaky, reset-by-subtraction, BETA=0.9, THR=1.0) applied on the
// C-tile layout.  reset computed from the *previous* membrane.
__device__ __forceinline__ v8f lif(v8f& mem, v8f acc, float bias) {
  v8f spk;
#pragma unroll
  for (int i = 0; i < 8; ++i) {
    float m0 = mem[i];
    float rst = (m0 > 1.0f) ? 1.0f : 0.0f;           // reset * THR
    float m = fmaf(0.9f, m0, acc[i] + bias) - rst;   // BETA*m + cur - reset
    mem[i] = m;
    spk[i] = (m > 1.0f) ? 1.0f : 0.0f;               // heaviside(m - THR)
  }
  return spk;
}

// C-tile (f32, row striped across lanes) -> f16 row-major staging in LDS.
// Lane holds col = lane&15, rows (lane<16 ? 0..7 : 8..15).
__device__ __forceinline__ void storeSpk(_Float16* base, int stride, int colbase,
                                         v8f spk, int ncols) {
  const int l = laneid();
  const int col = l & 15;
  if (col < ncols) {
    const int rb = (l < 16) ? 0 : 8;
#pragma unroll
    for (int i = 0; i < 8; ++i)
      base[(rb + i) * stride + colbase + col] = (_Float16)spk[i];
  }
}

// A-tile with K=16 from a 16-half-stride staging row (upper K zero).
__device__ __forceinline__ v16h ldsA16(const _Float16* rowp) {
  const int l = laneid();
  h8 lo = *(const h8*)(rowp + ((l < 16) ? 0 : 8));  // ds_load_b128
  h8 z = zero8h();
  return __builtin_shufflevector(lo, z, 0, 1, 2, 3, 4, 5, 6, 7, 8, 9, 10, 11, 12,
                                 13, 14, 15);
}

// A-tile with K=32 from a 64-half-stride staging row, K-chunk base kb (halves).
__device__ __forceinline__ v16h ldsA32(const _Float16* rowp, int kb) {
  const int l = laneid();
  const int off = kb + ((l < 16) ? 0 : 8);
  h8 lo = *(const h8*)(rowp + off);        // K = off .. off+7
  h8 hi = *(const h8*)(rowp + off + 16);   // K = off+16 .. off+23
  return __builtin_shufflevector(lo, hi, 0, 1, 2, 3, 4, 5, 6, 7, 8, 9, 10, 11, 12,
                                 13, 14, 15);
}

// A-tile built directly from a global input row (K = NF features, rest zero).
template <int NF>
__device__ __forceinline__ v16h inputA(const float* xrow) {
  v16h a = zero16h();
  if (laneid() < 16) {
#pragma unroll
    for (int f = 0; f < NF; ++f) a[f] = (_Float16)xrow[f];
  }
  return a;
}

__global__ __launch_bounds__(128) void snn_mpb_kernel(
    const float* __restrict__ left, const float* __restrict__ right,
    const float* __restrict__ ctx, const float* __restrict__ Wl1,
    const float* __restrict__ bl1, const float* __restrict__ Wl2,
    const float* __restrict__ bl2, const float* __restrict__ Wr1,
    const float* __restrict__ br1, const float* __restrict__ Wr2,
    const float* __restrict__ br2, const float* __restrict__ Wc,
    const float* __restrict__ bc, const float* __restrict__ Wf,
    const float* __restrict__ bf, const float* __restrict__ Wo,
    const float* __restrict__ bo, float* __restrict__ out) {
  // Per-wave staging: fused concat row [sl2(16)|sr2(16)|sc(8)|zeros(24)] and a
  // 16x16 scratch for layer-1 spikes.  All DS traffic is wave-private.
  __shared__ __align__(16) _Float16 fuseS[4][16][64];
  __shared__ __align__(16) _Float16 tmpS[4][16][16];

  const int wave = (int)(threadIdx.x >> 5);
  const int l = laneid();
  const int gw = (int)blockIdx.x * 4 + wave;   // one wave = 16 batch rows
  const int b0 = gw * 16;
  const int row = l & 15;
  const int n = l & 15;

  _Float16* fusew = &fuseS[wave][0][0];
  _Float16* tmpw = &tmpS[wave][0][0];
  _Float16* myrow16 = tmpw + row * 16;
  _Float16* myrow64 = fusew + row * 64;

  // Zero the K-pad region (cols 40..63) of the fused staging once.
  {
    const int c0 = 40 + ((l < 16) ? 0 : 12);
#pragma unroll
    for (int c = 0; c < 12; ++c) fusew[row * 64 + c0 + c] = (_Float16)0.0f;
  }

  // Weight B-tiles, resident in VGPRs across the whole T loop.
  const v16h Bl1 = makeB(Wl1, 16, 6, 0, 0);
  const v16h Bl2 = makeB(Wl2, 16, 16, 0, 0);
  const v16h Br1 = makeB(Wr1, 16, 6, 0, 0);
  const v16h Br2 = makeB(Wr2, 16, 16, 0, 0);
  const v16h Bcc = makeB(Wc, 8, 3, 0, 0);
  const v16h Bf00 = makeB(Wf, 32, 40, 0, 0);    // N 0..15, K 0..31
  const v16h Bf01 = makeB(Wf, 32, 40, 0, 32);   // N 0..15, K 32..39 (+pad)
  const v16h Bf10 = makeB(Wf, 32, 40, 16, 0);   // N 16..31, K 0..31
  const v16h Bf11 = makeB(Wf, 32, 40, 16, 32);  // N 16..31, K 32..39 (+pad)
  const v16h Boo = makeB(Wo, 4, 32, 0, 0);

  const float bl1v = bl1[n], bl2v = bl2[n], br1v = br1[n], br2v = br2[n];
  const float bcv = (n < 8) ? bc[n] : 0.0f;
  const float bf0v = bf[n], bf1v = bf[n + 16];
  const float bov = (n < 4) ? bo[n] : 0.0f;

  // Membrane states live in registers in C-tile layout.
  v8f ml1 = zero8f(), ml2 = zero8f(), mr1 = zero8f(), mr2 = zero8f();
  v8f mc = zero8f(), mf0 = zero8f(), mf1 = zero8f(), mo = zero8f();

  const float* xl = left + ((size_t)b0 + row) * 6;
  const float* xr = right + ((size_t)b0 + row) * 6;
  const float* xc = ctx + ((size_t)b0 + row) * 3;
  const size_t strideL = (size_t)BATCH * 6;
  const size_t strideC = (size_t)BATCH * 3;

  for (int t = 0; t < TSTEPS; ++t) {
    if ((t + 1 < TSTEPS) && (l < 16)) {
      __builtin_prefetch(xl + strideL, 0, 0);
      __builtin_prefetch(xr + strideL, 0, 0);
      __builtin_prefetch(xc + strideC, 0, 0);
    }

    // ---------------- left pathway: 6 -> 16 -> 16 ----------------
    v16h a = inputA<6>(xl);
    v8f acc = wmma16(a, Bl1, zero8f());
    v8f s = lif(ml1, acc, bl1v);
    storeSpk(tmpw, 16, 0, s, 16);
    a = ldsA16(myrow16);
    acc = wmma16(a, Bl2, zero8f());
    s = lif(ml2, acc, bl2v);
    storeSpk(fusew, 64, 0, s, 16);  // fused cols 0..15

    // ---------------- right pathway: 6 -> 16 -> 16 ----------------
    a = inputA<6>(xr);
    acc = wmma16(a, Br1, zero8f());
    s = lif(mr1, acc, br1v);
    storeSpk(tmpw, 16, 0, s, 16);
    a = ldsA16(myrow16);
    acc = wmma16(a, Br2, zero8f());
    s = lif(mr2, acc, br2v);
    storeSpk(fusew, 64, 16, s, 16);  // fused cols 16..31

    // ---------------- context pathway: 3 -> 8 ----------------
    a = inputA<3>(xc);
    acc = wmma16(a, Bcc, zero8f());
    s = lif(mc, acc, bcv);
    storeSpk(fusew, 64, 32, s, 8);  // fused cols 32..39 (cols>=8 stay zero-pad)

    // ---------------- fused layer: 40 -> 32 (K split 32+8pad, N split 16+16) --
    v16h a1 = ldsA32(myrow64, 0);
    v16h a2 = ldsA32(myrow64, 32);
    v8f f0 = wmma16(a2, Bf01, wmma16(a1, Bf00, zero8f()));
    v8f f1 = wmma16(a2, Bf11, wmma16(a1, Bf10, zero8f()));
    v8f sf0 = lif(mf0, f0, bf0v);
    v8f sf1 = lif(mf1, f1, bf1v);
    // Stage sf (16x32) over the consumed fused cols for the output layer.
    storeSpk(fusew, 64, 0, sf0, 16);
    storeSpk(fusew, 64, 16, sf1, 16);

    // ---------------- output layer: 32 -> 4 ----------------
    a1 = ldsA32(myrow64, 0);
    acc = wmma16(a1, Boo, zero8f());
    s = lif(mo, acc, bov);
    if (n < 4) {
      const int rb = (l < 16) ? 0 : 8;
      const size_t obase = ((size_t)t * BATCH + (size_t)b0) * 4;
#pragma unroll
      for (int i = 0; i < 8; ++i)
        out[obase + (size_t)(rb + i) * 4 + n] = s[i];
    }

    xl += strideL;
    xr += strideL;
    xc += strideC;
  }
}

extern "C" void kernel_launch(void* const* d_in, const int* in_sizes, int n_in,
                              void* d_out, int out_size, void* d_ws,
                              size_t ws_size, hipStream_t stream) {
  (void)in_sizes; (void)n_in; (void)d_ws; (void)ws_size; (void)out_size;
  const float* left = (const float*)d_in[0];
  const float* right = (const float*)d_in[1];
  const float* ctx = (const float*)d_in[2];
  const float* Wl1 = (const float*)d_in[3];
  const float* bl1 = (const float*)d_in[4];
  const float* Wl2 = (const float*)d_in[5];
  const float* bl2 = (const float*)d_in[6];
  const float* Wr1 = (const float*)d_in[7];
  const float* br1 = (const float*)d_in[8];
  const float* Wr2 = (const float*)d_in[9];
  const float* br2 = (const float*)d_in[10];
  const float* Wc = (const float*)d_in[11];
  const float* bc = (const float*)d_in[12];
  const float* Wf = (const float*)d_in[13];
  const float* bf = (const float*)d_in[14];
  const float* Wo = (const float*)d_in[15];
  const float* bo = (const float*)d_in[16];
  float* out = (float*)d_out;

  dim3 block(128);                 // 4 waves/block, each wave owns 16 batch rows
  dim3 grid(BATCH / (16 * 4));     // 64 blocks -> 256 waves total
  snn_mpb_kernel<<<grid, block, 0, stream>>>(left, right, ctx, Wl1, bl1, Wl2,
                                             bl2, Wr1, br1, Wr2, br2, Wc, bc,
                                             Wf, bf, Wo, bo, out);
}
